// Self_Attn_1709396984386
// MI455X (gfx1250) — compile-verified, matrix operand order checked by reading, not run on hardware
//
#include <hip/hip_runtime.h>
#include <hip/hip_bf16.h>

typedef __attribute__((ext_vector_type(16))) _Float16 v16h;
typedef __attribute__((ext_vector_type(8)))  _Float16 v8h;
typedef __attribute__((ext_vector_type(8)))  float    v8f;

#define CD_ 192
#define N_  9216
#define B_  2
#define KT_ 32          // key tile width per iteration

// ---------------------------------------------------------------------------
// Async global->LDS copy (gfx1250 GLOBAL_LOAD_ASYNC_TO_LDS_B128, ASYNCcnt).
// ---------------------------------------------------------------------------
#if defined(__has_builtin)
#  if __has_builtin(__builtin_amdgcn_global_load_async_to_lds_b128)
#    define HAVE_ASYNC_LDS 1
#  endif
#endif
#ifndef HAVE_ASYNC_LDS
#  define HAVE_ASYNC_LDS 0
#endif

typedef __attribute__((__vector_size__(16))) int i32x4;
typedef __attribute__((address_space(1))) i32x4* as1_i32x4p;
typedef __attribute__((address_space(3))) i32x4* as3_i32x4p;

__device__ __forceinline__ void copy16_to_lds(const _Float16* g, _Float16* l) {
#if HAVE_ASYNC_LDS
  __builtin_amdgcn_global_load_async_to_lds_b128(
      (as1_i32x4p)(uintptr_t)g, (as3_i32x4p)(uint32_t)(uintptr_t)l, 0, 0);
#else
  *(uint4*)l = *(const uint4*)g;
#endif
}

__device__ __forceinline__ void wait_async_lds() {
#if HAVE_ASYNC_LDS
#  if __has_builtin(__builtin_amdgcn_s_wait_asynccnt)
  __builtin_amdgcn_s_wait_asynccnt(0);
#  else
  asm volatile("s_wait_asynccnt 0x0" ::: "memory");
#  endif
#endif
}

// ---------------------------------------------------------------------------
// WMMA fragment loaders (CDNA5 wave32 layouts, CDNA5 ISA §7.12.2).
// A frag 16x32: lane l: row m=l&15, h=l>>4; elem i -> k=(i>>3)*16 + h*8 + (i&7)
// B frag 32x16: lane l: col n=l&15, same k mapping.
// => with "row-major in m (A) / n (B), k contiguous" sources, each fragment is
//    two contiguous 16B loads at k = h*8 and k = 16 + h*8.
// C/D frag 16x16 f32: lane l: col n=l&15; vgpr r -> row m = r + 8*h.
// ---------------------------------------------------------------------------
__device__ __forceinline__ v16h load_frag16(const _Float16* src, int ld) {
  int l = threadIdx.x & 31, r = l & 15, h = l >> 4;
  const _Float16* p = src + r * ld + (h << 3);
  v8h lo = *(const v8h*)p;          // k = h*8 .. h*8+7
  v8h hi = *(const v8h*)(p + 16);   // k = 16+h*8 ..
  return __builtin_shufflevector(lo, hi, 0, 1, 2, 3, 4, 5, 6, 7,
                                 8, 9, 10, 11, 12, 13, 14, 15);
}

__device__ __forceinline__ v16h load_a_f32v(const float* src, int lda) {
  int l = threadIdx.x & 31, m = l & 15, h = l >> 4;
  const float* p = src + m * lda + (h << 3);
  v8f lo = *(const v8f*)p;
  v8f hi = *(const v8f*)(p + 16);
  v16h a;
#pragma unroll
  for (int i = 0; i < 8; ++i) { a[i] = (_Float16)lo[i]; a[i + 8] = (_Float16)hi[i]; }
  return a;
}

__device__ __forceinline__ v8f wmma_f16(v16h a, v16h b, v8f c) {
  return __builtin_amdgcn_wmma_f32_16x16x32_f16(false, a, false, b,
                                                (short)0, c, false, false);
}

// ---------------------------------------------------------------------------
// Kernel 1: QKV projection.  out[o,n] = sum_c W[o,c]*x[c,n] + bias[o]
// Block = (mat, batch, 32 n-columns): stages x^T tile [32 n][192 c] f16 in LDS
// once, then 8 waves compute 24 output tiles (12 o-tiles x 2 n-halves).
// Q stored [N,CD] f16 (transposed), K stored [N,CD] f16 (transposed),
// V stored [CD,N] f16.
// ---------------------------------------------------------------------------
__global__ void __launch_bounds__(256)
qkv_proj_kernel(const float* __restrict__ x,
                const float* __restrict__ Wq, const float* __restrict__ bq,
                const float* __restrict__ Wk, const float* __restrict__ bk,
                const float* __restrict__ Wv, const float* __restrict__ bv,
                _Float16* __restrict__ qT, _Float16* __restrict__ kT,
                _Float16* __restrict__ vb) {
  __shared__ __align__(16) _Float16 xs[KT_ * CD_];   // [n][c], 12KB

  int blk = blockIdx.x;
  int nt32  = blk % (N_ / 32); blk /= (N_ / 32);
  int batch = blk % B_;        blk /= B_;
  int mat   = blk;                     // 0=Q 1=K 2=V (grid sized exactly)
  int n0 = nt32 * 32;

  const float* W    = (mat == 0) ? Wq : (mat == 1) ? Wk : Wv;
  const float* bias = (mat == 0) ? bq : (mat == 1) ? bk : bv;
  const float* xb   = x + (size_t)batch * CD_ * N_;
  int t = threadIdx.x;

  // ---- stage x^T tile: coalesced 32B reads along n, f16 transpose into LDS --
#pragma unroll
  for (int j = 0; j < 3; ++j) {
    int chunk = j * 256 + t;               // 768 chunks of 8 floats
    int c = chunk >> 2, noff = (chunk & 3) << 3;
    v8f xv = *(const v8f*)(xb + (size_t)c * N_ + n0 + noff);
#pragma unroll
    for (int e = 0; e < 8; ++e)
      xs[(noff + e) * CD_ + c] = (_Float16)xv[e];
  }
  __syncthreads();

  int l = t & 31, col = l & 15, half = l >> 4;
  int wib = t >> 5;

  // ---- 24 tiles: tau = wib + 8*s ; o-tile = tau>>1, n-half = tau&1 ----
#pragma unroll
  for (int s = 0; s < 3; ++s) {
    int tau = wib + (s << 3);
    int o0 = (tau >> 1) * 16;
    int nh = tau & 1;

    v8f acc = (v8f)(0.0f);
#pragma unroll
    for (int kc = 0; kc < CD_; kc += 32) {
      v16h a  = load_a_f32v(W + o0 * CD_ + kc, CD_);
      v16h bf = load_frag16(xs + nh * 16 * CD_ + kc, CD_);
      acc = wmma_f16(a, bf, acc);
    }

    int nbase = n0 + nh * 16;
#pragma unroll
    for (int r = 0; r < 8; ++r) {
      int o = o0 + r + 8 * half;
      int n = nbase + col;
      _Float16 hv = (_Float16)(acc[r] + bias[o]);
      if (mat == 0)      qT[(size_t)batch * N_ * CD_ + (size_t)n * CD_ + o] = hv;
      else if (mat == 1) kT[(size_t)batch * N_ * CD_ + (size_t)n * CD_ + o] = hv;
      else               vb[(size_t)batch * CD_ * N_ + (size_t)o * N_ + n] = hv;
    }
  }
}

// ---------------------------------------------------------------------------
// Cooperative K/V tile copy into LDS (async, double-buffered).
// K tile: [KT_][CD_] (ld=CD_), contiguous 12KB slice of kT at n0*CD_.
// V tile: [CD_][KT_] (ld=KT_), rows from vb[c*N + n0 + off].
// 12KB each => 3x 16B chunks per thread per tile (256 threads).
// ---------------------------------------------------------------------------
__device__ __forceinline__ void load_kv_tile(const _Float16* k, const _Float16* v,
                                             int n0, _Float16* kt, _Float16* vt,
                                             int t) {
#pragma unroll
  for (int j = 0; j < 3; ++j) {
    int idx = (j * 256 + t) * 8;          // half index, multiple of 8 (16B)
    copy16_to_lds(k + (size_t)n0 * CD_ + idx, kt + idx);
  }
#pragma unroll
  for (int j = 0; j < 3; ++j) {
    int idx = (j * 256 + t) * 8;
    int c = idx >> 5, off = idx & 31;     // V tile: row c, cols off..off+7
    copy16_to_lds(v + (size_t)c * N_ + n0 + off, vt + idx);
  }
}

// ---------------------------------------------------------------------------
// Kernel 2: flash attention.  Block = 8 waves = 128 queries; all waves share
// double-buffered K/V tiles staged in LDS via async copies.
// ---------------------------------------------------------------------------
__global__ void __launch_bounds__(256)
flash_attn_kernel(const _Float16* __restrict__ qT,
                  const _Float16* __restrict__ kTg,
                  const _Float16* __restrict__ vbg,
                  float* __restrict__ out) {
  __shared__ __align__(16) _Float16 kTile[2][KT_ * CD_];   // 2 x 12KB, [key][c]
  __shared__ __align__(16) _Float16 vTile[2][CD_ * KT_];   // 2 x 12KB, [c][key]
  __shared__ __align__(16) _Float16 pStage[8][16 * KT_];   // 8KB, [row][key]

  int t = threadIdx.x;
  int wib = t >> 5;
  int batch = blockIdx.x / (N_ / 128);
  int qblk  = blockIdx.x % (N_ / 128);
  int m0 = qblk * 128 + wib * 16;

  const _Float16* q = qT  + (size_t)batch * N_ * CD_;
  const _Float16* k = kTg + (size_t)batch * N_ * CD_;
  const _Float16* v = vbg + (size_t)batch * CD_ * N_;

  // Q fragments: 6 K-chunks of 32, resident for the whole sweep.
  v16h qa[6];
#pragma unroll
  for (int c = 0; c < 6; ++c) qa[c] = load_frag16(q + (size_t)m0 * CD_ + c * 32, CD_);

  v8f o_acc[12];
#pragma unroll
  for (int i = 0; i < 12; ++i) o_acc[i] = (v8f)(0.0f);

  float m_run[8], l_run[8];
#pragma unroll
  for (int r = 0; r < 8; ++r) { m_run[r] = -3.0e38f; l_run[r] = 0.0f; }

  int l = t & 31, col = l & 15, half = l >> 4;
  _Float16* myp = &pStage[wib][0];

  // Prologue: stage first tile into buffer 0.
  load_kv_tile(k, v, 0, kTile[0], vTile[0], t);
  wait_async_lds();
  __syncthreads();

  for (int it = 0, n0 = 0; n0 < N_; n0 += KT_, ++it) {
    int cur = it & 1;
    if (n0 + KT_ < N_)                          // prefetch next tile into buf^1
      load_kv_tile(k, v, n0 + KT_, kTile[cur ^ 1], vTile[cur ^ 1], t);

    const _Float16* kt = kTile[cur];
    const _Float16* vt = vTile[cur];

    // ---- S = Q K^T for this 16x32 key tile (two 16x16 column halves) ----
    // B frag (keys 0-15):  kt[key*CD_ + c0 + k] ; (keys 16-31): +16*CD_
    v8f s0 = (v8f)(0.0f), s1 = (v8f)(0.0f);
#pragma unroll
    for (int c = 0; c < 6; ++c) {
      v16h b0 = load_frag16(kt + c * 32, CD_);
      v16h b1 = load_frag16(kt + 16 * CD_ + c * 32, CD_);
      s0 = wmma_f16(qa[c], b0, s0);
      s1 = wmma_f16(qa[c], b1, s1);
    }

    // ---- row max over 32 columns (shfl within 16-lane row groups) ----
    float rmax[8];
#pragma unroll
    for (int r = 0; r < 8; ++r) rmax[r] = fmaxf(s0[r], s1[r]);
#pragma unroll
    for (int off = 8; off >= 1; off >>= 1)
#pragma unroll
      for (int r = 0; r < 8; ++r) rmax[r] = fmaxf(rmax[r], __shfl_xor(rmax[r], off));

    float alpha[8], rsum[8];
#pragma unroll
    for (int r = 0; r < 8; ++r) {
      float mn = fmaxf(m_run[r], rmax[r]);
      alpha[r] = __expf(m_run[r] - mn);
      m_run[r] = mn;
    }

    // ---- P = exp(S - m): stage row-major 16x32 f16 tile in private LDS ----
#pragma unroll
    for (int r = 0; r < 8; ++r) {
      float p0 = __expf(s0[r] - m_run[r]);
      float p1 = __expf(s1[r] - m_run[r]);
      int row = r + 8 * half;
      myp[row * KT_ + col]      = (_Float16)p0;
      myp[row * KT_ + col + 16] = (_Float16)p1;
      rsum[r] = p0 + p1;
    }
#pragma unroll
    for (int off = 8; off >= 1; off >>= 1)
#pragma unroll
      for (int r = 0; r < 8; ++r) rsum[r] += __shfl_xor(rsum[r], off);
#pragma unroll
    for (int r = 0; r < 8; ++r) l_run[r] = l_run[r] * alpha[r] + rsum[r];

    // Re-shape P (C-layout -> A-fragment) via same-wave LDS round-trip.
    v16h pa = load_frag16(myp, KT_);

    // ---- rescale accumulators, then out += P . V^T (12 channel tiles) ----
    // B frag for channel tile i: vt[(i*16 + c)*KT_ + key], k=key contiguous.
#pragma unroll
    for (int i = 0; i < 12; ++i)
#pragma unroll
      for (int r = 0; r < 8; ++r) o_acc[i][r] *= alpha[r];
#pragma unroll
    for (int i = 0; i < 12; ++i) {
      v16h bvf = load_frag16(vt + i * 16 * KT_, KT_);
      o_acc[i] = wmma_f16(pa, bvf, o_acc[i]);
    }

    wait_async_lds();          // next tile resident before buffers swap
    __syncthreads();
  }

  // ---- epilogue: out[c, m] = acc / l ----
#pragma unroll
  for (int i = 0; i < 12; ++i)
#pragma unroll
    for (int r = 0; r < 8; ++r) {
      int c = i * 16 + col;
      int m = m0 + r + 8 * half;
      out[(size_t)batch * CD_ * N_ + (size_t)c * N_ + m] = o_acc[i][r] / l_run[r];
    }
}

// ---------------------------------------------------------------------------
extern "C" void kernel_launch(void* const* d_in, const int* in_sizes, int n_in,
                              void* d_out, int out_size, void* d_ws, size_t ws_size,
                              hipStream_t stream) {
  (void)in_sizes; (void)n_in; (void)out_size; (void)ws_size;
  const float* x  = (const float*)d_in[0];
  const float* Wq = (const float*)d_in[1];
  const float* bq = (const float*)d_in[2];
  const float* Wk = (const float*)d_in[3];
  const float* bk = (const float*)d_in[4];
  const float* Wv = (const float*)d_in[5];
  const float* bv = (const float*)d_in[6];
  float* out = (float*)d_out;

  const size_t bufElems = (size_t)B_ * N_ * CD_;         // per q/k/v f16 buffer
  _Float16* qT = (_Float16*)d_ws;
  _Float16* kT = qT + bufElems;
  _Float16* vb = kT + bufElems;

  // Projection: 3 mats * B * (N/32) blocks; each computes 12x2 output tiles.
  qkv_proj_kernel<<<3 * B_ * (N_ / 32), 256, 0, stream>>>(x, Wq, bq, Wk, bk,
                                                          Wv, bv, qT, kT, vb);

  // Attention: B * (N/128) blocks of 8 query-tile waves.
  flash_attn_kernel<<<B_ * (N_ / 128), 256, 0, stream>>>(qT, kT, vb, out);
}
